// SlotAttention_30382598652357
// MI455X (gfx1250) — matrix operand order repaired and checked by reading, not compiled, not run on hardware
//
#include <hip/hip_runtime.h>
#include <hip/hip_bf16.h>

// ---------------------------------------------------------------------------
// Slot Attention on MI455X (gfx1250), wave32 + WMMA bf16 (f32 accumulate).
//
// Pipeline:
//   prep:   compose linear stack -> W_h(96x128 padded), pack W_h/wk/wv into
//           WMMA-B fragment layout (bf16)
//   init:   slots = mean(mu) + std(sigma)*gauss(hash)
//   hkv:    fused  h = [in,emb]@W_h+b_h  -> LayerNorm -> k=..@wk, v=..@wv
//           k stored bf16 row-major [b][n][d]; v stored bf16 transposed [b][d][n]
//   3x iterations:
//     qproj: q = LN(slots)@wq+bq, packed as WMMA-B fragments (slots padded to 16)
//     zero:  U,R = 0
//     attn:  dots = k@q^T (WMMA), softmax over slots (per-lane), a=softmax+EPS,
//            U += a^T@v (WMMA via LDS transpose of a), R += sum(a); atomics
//     finish: updates=U/R; GRUCell; LN; 2-layer MLP residual; write slots
// ---------------------------------------------------------------------------

#define BB 8
#define NN 32768
#define DD 128

typedef float v8f __attribute__((ext_vector_type(8)));
typedef __bf16 v16bf __attribute__((ext_vector_type(16)));

union BFrag {
  v16bf v;
  __bf16 h[16];
  uint4  q[2];
};

__device__ inline v8f vzero8() {
  v8f z = {0.f, 0.f, 0.f, 0.f, 0.f, 0.f, 0.f, 0.f};
  return z;
}

__device__ inline v8f wmma_bf16(const BFrag& a, const BFrag& b, v8f c) {
  // D = A(16x32 bf16) x B(32x16 bf16) + C(16x16 f32)
  return __builtin_amdgcn_wmma_f32_16x16x32_bf16(
      false, a.v, false, b.v, (short)0, c, false, false);
}

// ---------------------------------------------------------------------------
// Kernel A: weight preparation (one block).
// ---------------------------------------------------------------------------
__global__ __launch_bounds__(256) void prep_kernel(
    const float* __restrict__ w0, const float* __restrict__ b0,
    const float* __restrict__ w1, const float* __restrict__ b1,
    const float* __restrict__ w2, const float* __restrict__ b2,
    const float* __restrict__ w3, const float* __restrict__ b3,
    const float* __restrict__ w4, const float* __restrict__ b4,
    const float* __restrict__ w5, const float* __restrict__ b5,
    const float* __restrict__ w7, const float* __restrict__ b7,
    const float* __restrict__ wk, const float* __restrict__ wvv,
    float* __restrict__ Whf, float* __restrict__ bh,
    __bf16* __restrict__ Whp, __bf16* __restrict__ wkp,
    __bf16* __restrict__ wvp)
{
  __shared__ float Ma[3 * 64];
  __shared__ float Mb[3 * 32];
  __shared__ float ba[64];
  __shared__ float bb2[32];
  const int t = threadIdx.x;

  for (int i = t; i < 192; i += 256) Ma[i] = w0[i];
  for (int i = t; i < 64; i += 256) ba[i] = b0[i];
  __syncthreads();

  const float* Ws[5] = {w1, w2, w3, w4, w5};
  const float* Bs[5] = {b1, b2, b3, b4, b5};
  int K = 64;
  for (int st = 0; st < 5; ++st) {
    const float* W = Ws[st];
    const float* Bc = Bs[st];
    if (t < 96) {
      int i = t / 32, j = t % 32;
      float a = 0.f;
      for (int k2 = 0; k2 < K; ++k2) a += Ma[i * K + k2] * W[k2 * 32 + j];
      Mb[i * 32 + j] = a;
    } else if (t < 128) {
      int j = t - 96;
      float a = Bc[j];
      for (int k2 = 0; k2 < K; ++k2) a += ba[k2] * W[k2 * 32 + j];
      bb2[j] = a;
    }
    __syncthreads();
    if (t < 96) Ma[t] = Mb[t];
    if (t < 32) ba[t] = bb2[t];
    K = 32;
    __syncthreads();
  }
  // Ma = Weff (3x32), ba = beff (32)

  // W_h: rows 0..2 = w7[i]+Weff@w7[66:98]; rows 3..65 = w7[3+j]; 66..95 = 0
  for (int idx = t; idx < 96 * 128; idx += 256) {
    int rr = idx >> 7, c = idx & 127;
    float s = 0.f;
    if (rr < 3) {
      s = w7[rr * 128 + c];
      for (int k2 = 0; k2 < 32; ++k2) s += Ma[rr * 32 + k2] * w7[(66 + k2) * 128 + c];
    } else if (rr < 66) {
      s = w7[rr * 128 + c];
    }
    Whf[idx] = s;
  }
  for (int c = t; c < 128; c += 256) {
    float a = b7[c];
    for (int k2 = 0; k2 < 32; ++k2) a += ba[k2] * w7[(66 + k2) * 128 + c];
    bh[c] = a;
  }
  __syncthreads();

  // Pack W_h into WMMA-B fragment layout: Whp[((ct*3+ks)*32+lane)*16+e]
  // element e of lane: K = (e&7) + 8*(lane>>4) + 16*(e>>3) + 32*ks, col = ct*16 + (lane&15)
  for (int idx = t; idx < 8 * 3 * 32 * 16; idx += 256) {
    int e = idx & 15, ln2 = (idx >> 4) & 31, rest = idx >> 9;
    int ks = rest % 3, ct = rest / 3;
    int Kk = (e & 7) + 8 * (ln2 >> 4) + 16 * (e >> 3) + 32 * ks;
    int col = ct * 16 + (ln2 & 15);
    Whp[idx] = (__bf16)Whf[Kk * 128 + col];
  }
  // Pack wk / wv (K=128 -> 4 ksteps)
  for (int idx = t; idx < 8 * 4 * 32 * 16; idx += 256) {
    int e = idx & 15, ln2 = (idx >> 4) & 31, rest = idx >> 9;
    int ks = rest & 3, ct = rest >> 2;
    int Kk = (e & 7) + 8 * (ln2 >> 4) + 16 * (e >> 3) + 32 * ks;
    int col = ct * 16 + (ln2 & 15);
    wkp[idx] = (__bf16)wk[Kk * 128 + col];
    wvp[idx] = (__bf16)wvv[Kk * 128 + col];
  }
}

// ---------------------------------------------------------------------------
// Kernel C: slot initialization (one block, 128 threads).
// ---------------------------------------------------------------------------
__device__ inline unsigned hash_u(unsigned x) {
  x ^= x >> 17; x *= 0xed5ad4bbu;
  x ^= x >> 11; x *= 0xac4c1b51u;
  x ^= x >> 15; x *= 0x31848babu;
  x ^= x >> 14;
  return x;
}

__device__ inline float gaussh(int i) {
  unsigned u1 = hash_u(0x9E3779B9u ^ (unsigned)i);
  unsigned u2 = hash_u(0x85EBCA6Bu + (unsigned)i * 0x27d4eb2du);
  float f1 = (u1 >> 8) * (1.0f / 16777216.0f) + 1e-7f;
  float f2 = (u2 >> 8) * (1.0f / 16777216.0f);
  return sqrtf(-2.f * __logf(f1)) * __cosf(6.28318530718f * f2);
}

__global__ __launch_bounds__(128) void initslots_kernel(
    const float* __restrict__ mu, const float* __restrict__ sg,
    float* __restrict__ slots)
{
  __shared__ float red[12];
  const int t = threadIdx.x;
  float m_ = mu[t], s_ = sg[t];
  float a = m_, b2 = s_;
  for (int m = 1; m < 32; m <<= 1) {
    a += __shfl_xor(a, m, 32);
    b2 += __shfl_xor(b2, m, 32);
  }
  if ((t & 31) == 0) { red[t >> 5] = a; red[4 + (t >> 5)] = b2; }
  __syncthreads();
  float mum = (red[0] + red[1] + red[2] + red[3]) * (1.f / 128.f);
  float sgm = (red[4] + red[5] + red[6] + red[7]) * (1.f / 128.f);
  float d = s_ - sgm;
  float dsq = d * d;
  for (int m = 1; m < 32; m <<= 1) dsq += __shfl_xor(dsq, m, 32);
  if ((t & 31) == 0) red[8 + (t >> 5)] = dsq;
  __syncthreads();
  float ssq = red[8] + red[9] + red[10] + red[11];
  // torch.std (ddof=1) over the (B,S,D)=8192-element broadcast: 64 copies/value
  float stdv = sqrtf(64.f * ssq / 8191.f);
  for (int idx = t; idx < BB * 8 * DD; idx += 128)
    slots[idx] = mum + stdv * gaussh(idx);
}

// ---------------------------------------------------------------------------
// Kernel B: fused h = X@W_h+b_h -> LayerNorm -> k,v projections.
// 2048 blocks x 256 threads (8 waves); each block: 128 rows.
// ---------------------------------------------------------------------------
__global__ __launch_bounds__(256) void hkv_kernel(
    const float* __restrict__ inp, const float* __restrict__ emb,
    const __bf16* __restrict__ Whp, const float* __restrict__ bh,
    const __bf16* __restrict__ wkp, const float* __restrict__ bk,
    const __bf16* __restrict__ wvp, const float* __restrict__ bvv,
    const float* __restrict__ g_ln, const float* __restrict__ b_ln,
    __bf16* __restrict__ kbf, __bf16* __restrict__ vT)
{
  __shared__ __attribute__((aligned(16))) __bf16 Ain[128 * 96];   // 24 KB
  __shared__ __attribute__((aligned(16))) __bf16 hN[128 * 136];   // 34 KB
  const int tid = threadIdx.x;
  const int lane = tid & 31;
  const int rt = tid >> 5;            // wave id = row tile 0..7
  const int half = lane >> 4;
  const int l15 = lane & 15;
  const long long g0 = (long long)blockIdx.x * 128;
  const int b = (int)(g0 >> 15);
  const int n0 = (int)(g0 & (NN - 1));

  // stage input tile [128 x 96] (cols 0..2 inputs, 3..65 embed, 66..95 zero)
  for (int idx = tid; idx < 128 * 96; idx += 256) {
    int r = idx / 96, c = idx % 96;
    long long row = (long long)b * NN + n0 + r;
    float x = 0.f;
    if (c < 3) x = inp[row * 3 + c];
    else if (c < 66) x = emb[row * 63 + (c - 3)];
    Ain[r * 96 + c] = (__bf16)x;
  }
  __syncthreads();

  const int rowA = rt * 16 + l15;     // A-operand row (M) for this lane

  // ---- stage 1: h tile = X @ W_h  (K = 96 -> 3 ksteps) ----
  v8f acc[8];
#pragma unroll
  for (int ct = 0; ct < 8; ++ct) acc[ct] = vzero8();
#pragma unroll
  for (int ks = 0; ks < 3; ++ks) {
    BFrag af;
    int kb = 32 * ks + 8 * half;
    af.q[0] = *(const uint4*)&Ain[rowA * 96 + kb];
    af.q[1] = *(const uint4*)&Ain[rowA * 96 + kb + 16];
#pragma unroll
    for (int ct = 0; ct < 8; ++ct) {
      BFrag bf_;
      const __bf16* p = Whp + ((size_t)(ct * 3 + ks) * 32 + lane) * 16;
      bf_.q[0] = *(const uint4*)p;
      bf_.q[1] = *(const uint4*)(p + 8);
      acc[ct] = wmma_bf16(af, bf_, acc[ct]);
    }
  }
  // bias
#pragma unroll
  for (int ct = 0; ct < 8; ++ct) {
    float bb = bh[ct * 16 + l15];
#pragma unroll
    for (int r = 0; r < 8; ++r) acc[ct][r] += bb;
  }
  // LayerNorm per row; row m = rt*16 + r + 8*half, its 128 values live in
  // the 16 lanes of this half across the 8 ct accumulators.
#pragma unroll
  for (int r = 0; r < 8; ++r) {
    float s = 0.f, s2 = 0.f;
#pragma unroll
    for (int ct = 0; ct < 8; ++ct) { float x = acc[ct][r]; s += x; s2 += x * x; }
    for (int m = 1; m < 16; m <<= 1) {
      s += __shfl_xor(s, m, 32);
      s2 += __shfl_xor(s2, m, 32);
    }
    float mean = s * (1.f / 128.f);
    float var = s2 * (1.f / 128.f) - mean * mean;
    float rstd = rsqrtf(var + 1e-5f);
    int mrow = rt * 16 + r + 8 * half;
#pragma unroll
    for (int ct = 0; ct < 8; ++ct) {
      int col = ct * 16 + l15;
      float x = (acc[ct][r] - mean) * rstd * g_ln[col] + b_ln[col];
      hN[mrow * 136 + col] = (__bf16)x;
    }
  }
  __syncthreads();

  // ---- stage 2a: k = hN @ wk ----
#pragma unroll
  for (int ct = 0; ct < 8; ++ct) acc[ct] = vzero8();
#pragma unroll
  for (int ks = 0; ks < 4; ++ks) {
    BFrag af;
    int kb = 32 * ks + 8 * half;
    af.q[0] = *(const uint4*)&hN[rowA * 136 + kb];
    af.q[1] = *(const uint4*)&hN[rowA * 136 + kb + 16];
#pragma unroll
    for (int ct = 0; ct < 8; ++ct) {
      BFrag bf_;
      const __bf16* p = wkp + ((size_t)(ct * 4 + ks) * 32 + lane) * 16;
      bf_.q[0] = *(const uint4*)p;
      bf_.q[1] = *(const uint4*)(p + 8);
      acc[ct] = wmma_bf16(af, bf_, acc[ct]);
    }
  }
#pragma unroll
  for (int ct = 0; ct < 8; ++ct) {
    int col = ct * 16 + l15;
    float bb = bk[col];
#pragma unroll
    for (int r = 0; r < 8; ++r) {
      int mrow = rt * 16 + r + 8 * half;
      kbf[((size_t)b * NN + n0 + mrow) * DD + col] = (__bf16)(acc[ct][r] + bb);
    }
  }

  // ---- stage 2b: v = hN @ wv, stored transposed vT[b][d][n] ----
#pragma unroll
  for (int ct = 0; ct < 8; ++ct) acc[ct] = vzero8();
#pragma unroll
  for (int ks = 0; ks < 4; ++ks) {
    BFrag af;
    int kb = 32 * ks + 8 * half;
    af.q[0] = *(const uint4*)&hN[rowA * 136 + kb];
    af.q[1] = *(const uint4*)&hN[rowA * 136 + kb + 16];
#pragma unroll
    for (int ct = 0; ct < 8; ++ct) {
      BFrag bf_;
      const __bf16* p = wvp + ((size_t)(ct * 4 + ks) * 32 + lane) * 16;
      bf_.q[0] = *(const uint4*)p;
      bf_.q[1] = *(const uint4*)(p + 8);
      acc[ct] = wmma_bf16(af, bf_, acc[ct]);
    }
  }
#pragma unroll
  for (int ct = 0; ct < 8; ++ct) {
    int col = ct * 16 + l15;
    float bb = bvv[col];
    __attribute__((aligned(16))) __bf16 pk[8];
#pragma unroll
    for (int r = 0; r < 8; ++r) pk[r] = (__bf16)(acc[ct][r] + bb);
    // consecutive r == consecutive positions -> one b128 store
    size_t base = ((size_t)b * DD + col) * NN + n0 + rt * 16 + 8 * half;
    *(uint4*)&vT[base] = *(const uint4*)pk;
  }
}

// ---------------------------------------------------------------------------
// Kernel D: q = LN(slots)@wq + bq, packed as WMMA-B fragments (16 slot cols,
// rows 8..15 zero). One block per batch.
// ---------------------------------------------------------------------------
__global__ __launch_bounds__(128) void qproj_kernel(
    const float* __restrict__ slots, const float* __restrict__ g,
    const float* __restrict__ bb, const float* __restrict__ wq,
    const float* __restrict__ bq, __bf16* __restrict__ qTp)
{
  __shared__ float lnrow[8 * 128];
  __shared__ float qbuf[8 * 128];
  __shared__ float stats[16];
  const int b = blockIdx.x, t = threadIdx.x;
  const float* sl = slots + (size_t)b * 8 * DD;
  if (t < 8) {
    float s = 0.f, s2 = 0.f;
    for (int d = 0; d < 128; ++d) { float x = sl[t * 128 + d]; s += x; s2 += x * x; }
    float mean = s * (1.f / 128.f);
    float var = s2 * (1.f / 128.f) - mean * mean;
    stats[t] = mean;
    stats[8 + t] = rsqrtf(var + 1e-5f);
  }
  __syncthreads();
  for (int idx = t; idx < 1024; idx += 128) {
    int s = idx >> 7, d = idx & 127;
    lnrow[idx] = (sl[idx] - stats[s]) * stats[8 + s] * g[d] + bb[d];
  }
  __syncthreads();
  for (int idx = t; idx < 1024; idx += 128) {
    int s = idx >> 7, c = idx & 127;
    float a = bq[c];
    for (int k2 = 0; k2 < 128; ++k2) a += lnrow[s * 128 + k2] * wq[k2 * 128 + c];
    qbuf[idx] = a;
  }
  __syncthreads();
  // pack: qTp[b][((ks*32+lane)*16+e)] = q[slot = lane&15][K = (e&7)+8*(lane>>4)+16*(e>>3)+32*ks]
  for (int idx = t; idx < 2048; idx += 128) {
    int e = idx & 15, ln2 = (idx >> 4) & 31, ks = idx >> 9;
    int slot2 = ln2 & 15;
    int Kk = (e & 7) + 8 * (ln2 >> 4) + 16 * (e >> 3) + 32 * ks;
    float val = (slot2 < 8) ? qbuf[slot2 * 128 + Kk] : 0.f;
    qTp[(size_t)b * 2048 + idx] = (__bf16)val;
  }
}

__global__ __launch_bounds__(256) void zero_kernel(float* __restrict__ U,
                                                   float* __restrict__ R)
{
  int i = blockIdx.x * 256 + threadIdx.x;
  if (i < BB * 16 * DD) U[i] = 0.f;
  if (i < BB * 16) R[i] = 0.f;
}

// ---------------------------------------------------------------------------
// Kernel E: attention pass. 64 blocks x 4 waves; each wave streams 1024
// positions: dots WMMA -> per-lane softmax over slots -> LDS transpose ->
// updates WMMA; atomically merges partial U (16x128) and R.
// ---------------------------------------------------------------------------
__global__ __launch_bounds__(128) void attn_kernel(
    const __bf16* __restrict__ kbf, const __bf16* __restrict__ vT,
    const __bf16* __restrict__ qTp, float* __restrict__ U,
    float* __restrict__ R)
{
  __shared__ __attribute__((aligned(16))) __bf16 aT[4][16 * 32];
  const int tid = threadIdx.x, lane = tid & 31, wv_ = tid >> 5;
  const int gw = blockIdx.x * 4 + wv_;
  const int b = gw >> 5;
  const int posbase = (gw & 31) * 1024;
  const int half = lane >> 4, l15 = lane & 15;
  const float scale = 0.08838834764831845f;  // 128^-0.5

  BFrag qf[4];
#pragma unroll
  for (int ks = 0; ks < 4; ++ks) {
    const __bf16* p = qTp + (size_t)b * 2048 + ((size_t)ks * 32 + lane) * 16;
    qf[ks].q[0] = *(const uint4*)p;
    qf[ks].q[1] = *(const uint4*)(p + 8);
  }

  v8f uacc[8];
#pragma unroll
  for (int ct = 0; ct < 8; ++ct) uacc[ct] = vzero8();
  float rAcc = 0.f;

  for (int step = 0; step < 32; ++step) {
    const int pos0 = posbase + step * 32;
#pragma unroll
    for (int pt = 0; pt < 2; ++pt) {
      v8f dacc = vzero8();
      const int row = pos0 + pt * 16 + l15;
      const __bf16* krow = kbf + ((size_t)b * NN + row) * DD;
#pragma unroll
      for (int ks = 0; ks < 4; ++ks) {
        BFrag af;
        int kb = 32 * ks + 8 * half;
        af.q[0] = *(const uint4*)&krow[kb];
        af.q[1] = *(const uint4*)&krow[kb + 16];
        dacc = wmma_bf16(af, qf[ks], dacc);
      }
      // dacc: M = position (r + 8*half), N = slot (l15). Softmax over slots:
      // the 8 real slot values for one position live in lanes {h*16+0..7},
      // reduced with shfl masks 1,2,4 (stays inside the 8-lane group).
#pragma unroll
      for (int r = 0; r < 8; ++r) {
        float x = dacc[r] * scale;
        float mx = x;
        mx = fmaxf(mx, __shfl_xor(mx, 1, 32));
        mx = fmaxf(mx, __shfl_xor(mx, 2, 32));
        mx = fmaxf(mx, __shfl_xor(mx, 4, 32));
        float e = __expf(x - mx);
        float s = e;
        s += __shfl_xor(s, 1, 32);
        s += __shfl_xor(s, 2, 32);
        s += __shfl_xor(s, 4, 32);
        float a = e / s + 1e-8f;       // softmax over slots + EPS
        rAcc += a;                     // meaningful only where l15 < 8
        aT[wv_][l15 * 32 + pt * 16 + r + 8 * half] = (__bf16)a;
      }
    }
    // updates: U[slot][d] += sum_pos a^T[slot][pos] * v[pos][d]; one K=32 step
    BFrag af2;
    {
      int kb2 = 8 * half;
      af2.q[0] = *(const uint4*)&aT[wv_][l15 * 32 + kb2];
      af2.q[1] = *(const uint4*)&aT[wv_][l15 * 32 + kb2 + 16];
    }
#pragma unroll
    for (int ct = 0; ct < 8; ++ct) {
      BFrag bf_;
      const __bf16* p = vT + ((size_t)b * DD + ct * 16 + l15) * NN + pos0 + 8 * half;
      bf_.q[0] = *(const uint4*)p;
      bf_.q[1] = *(const uint4*)(p + 16);
      uacc[ct] = wmma_bf16(af2, bf_, uacc[ct]);
    }
  }

  // merge partials: uacc element r of lane: M = slot = r + 8*half, N = d.
  if (half == 0) {
#pragma unroll
    for (int ct = 0; ct < 8; ++ct)
#pragma unroll
      for (int r = 0; r < 8; ++r)
        atomicAdd(&U[((size_t)b * 16 + r) * DD + ct * 16 + l15], uacc[ct][r]);
  }
  if (l15 < 8) atomicAdd(&R[b * 16 + l15], rAcc);
}

// ---------------------------------------------------------------------------
// Kernel F: updates = U/R; GRUCell; LN_ff; 2-layer MLP residual.
// 64 blocks (one per (b,slot)) x 128 threads (one per d).
// ---------------------------------------------------------------------------
__global__ __launch_bounds__(128) void finish_kernel(
    float* __restrict__ slots, const float* __restrict__ U,
    const float* __restrict__ R, const float* __restrict__ wih,
    const float* __restrict__ whh, const float* __restrict__ bih,
    const float* __restrict__ bhh, const float* __restrict__ lng,
    const float* __restrict__ lnb, const float* __restrict__ m1w,
    const float* __restrict__ m1b, const float* __restrict__ m2w,
    const float* __restrict__ m2b, float* __restrict__ outp, int writeOut)
{
  __shared__ float xg[128], hp[128], gi[384], gh[384];
  __shared__ float lnbuf[128], m1buf[128];
  __shared__ float redbuf[16];
  const int blk = blockIdx.x, b = blk >> 3, s = blk & 7, t = threadIdx.x;

  const float rinv = 1.f / R[b * 16 + s];
  xg[t] = U[((size_t)b * 16 + s) * DD + t] * rinv;
  hp[t] = slots[((size_t)(b * 8 + s)) * DD + t];
  __syncthreads();

  for (int j = t; j < 384; j += 128) {
    float a1 = bih[j], a2 = bhh[j];
    for (int k2 = 0; k2 < 128; ++k2) {
      a1 += wih[j * 128 + k2] * xg[k2];
      a2 += whh[j * 128 + k2] * hp[k2];
    }
    gi[j] = a1;
    gh[j] = a2;
  }
  __syncthreads();

  float rr = 1.f / (1.f + __expf(-(gi[t] + gh[t])));
  float zz = 1.f / (1.f + __expf(-(gi[128 + t] + gh[128 + t])));
  float nn = tanhf(gi[256 + t] + rr * gh[256 + t]);
  float sv = (1.f - zz) * nn + zz * hp[t];

  // LayerNorm over the 128-dim slot vector
  float x = sv, x2 = sv * sv;
  for (int m = 1; m < 32; m <<= 1) {
    x += __shfl_xor(x, m, 32);
    x2 += __shfl_xor(x2, m, 32);
  }
  if ((t & 31) == 0) { redbuf[t >> 5] = x; redbuf[8 + (t >> 5)] = x2; }
  __syncthreads();
  float s1 = redbuf[0] + redbuf[1] + redbuf[2] + redbuf[3];
  float s2 = redbuf[8] + redbuf[9] + redbuf[10] + redbuf[11];
  float mean = s1 * (1.f / 128.f);
  float var = s2 * (1.f / 128.f) - mean * mean;
  float rstd = rsqrtf(var + 1e-5f);
  lnbuf[t] = (sv - mean) * rstd * lng[t] + lnb[t];
  __syncthreads();

  float a1 = m1b[t];
  for (int k2 = 0; k2 < 128; ++k2) a1 += lnbuf[k2] * m1w[k2 * 128 + t];
  m1buf[t] = a1;
  __syncthreads();
  float a2 = m2b[t];
  for (int k2 = 0; k2 < 128; ++k2) a2 += m1buf[k2] * m2w[k2 * 128 + t];

  float outv = sv + a2;
  slots[((size_t)(b * 8 + s)) * DD + t] = outv;
  if (writeOut) outp[((size_t)(b * 8 + s)) * DD + t] = outv;
}

// ---------------------------------------------------------------------------
extern "C" void kernel_launch(void* const* d_in, const int* in_sizes, int n_in,
                              void* d_out, int out_size, void* d_ws,
                              size_t ws_size, hipStream_t stream)
{
  const float* inp = (const float*)d_in[0];
  const float* emb = (const float*)d_in[1];
  const float* mu = (const float*)d_in[2];
  const float* sg = (const float*)d_in[3];
  const float* w0 = (const float*)d_in[4];  const float* b0 = (const float*)d_in[5];
  const float* w1 = (const float*)d_in[6];  const float* b1 = (const float*)d_in[7];
  const float* w2 = (const float*)d_in[8];  const float* b2 = (const float*)d_in[9];
  const float* w3 = (const float*)d_in[10]; const float* b3 = (const float*)d_in[11];
  const float* w4 = (const float*)d_in[12]; const float* b4 = (const float*)d_in[13];
  const float* w5 = (const float*)d_in[14]; const float* b5 = (const float*)d_in[15];
  const float* w7 = (const float*)d_in[16]; const float* b7 = (const float*)d_in[17];
  const float* wq = (const float*)d_in[18]; const float* bq = (const float*)d_in[19];
  const float* wk = (const float*)d_in[20]; const float* bk = (const float*)d_in[21];
  const float* wv = (const float*)d_in[22]; const float* bv = (const float*)d_in[23];
  const float* ling = (const float*)d_in[24]; const float* linb = (const float*)d_in[25];
  const float* lsg = (const float*)d_in[26];  const float* lsb = (const float*)d_in[27];
  const float* lfg = (const float*)d_in[28];  const float* lfb = (const float*)d_in[29];
  const float* wih = (const float*)d_in[30];  const float* whh = (const float*)d_in[31];
  const float* bih = (const float*)d_in[32];  const float* bhh = (const float*)d_in[33];
  const float* m1w = (const float*)d_in[34];  const float* m1b = (const float*)d_in[35];
  const float* m2w = (const float*)d_in[36];  const float* m2b = (const float*)d_in[37];
  float* outp = (float*)d_out;

  char* base = (char*)d_ws;
  size_t off = 0;
  auto take = [&](size_t bytes) -> char* {
    char* p = base + off;
    off = (off + bytes + 255) & ~(size_t)255;
    return p;
  };
  __bf16* Whp = (__bf16*)take(8 * 3 * 32 * 16 * 2);
  __bf16* wkp = (__bf16*)take(8 * 4 * 32 * 16 * 2);
  __bf16* wvp = (__bf16*)take(8 * 4 * 32 * 16 * 2);
  float* Whf = (float*)take(96 * 128 * 4);
  float* bh = (float*)take(128 * 4);
  float* slots = (float*)take((size_t)BB * 8 * DD * 4);
  __bf16* qTp = (__bf16*)take((size_t)BB * 2048 * 2);
  float* U = (float*)take((size_t)BB * 16 * DD * 4);
  float* R = (float*)take((size_t)BB * 16 * 4);
  __bf16* kbf = (__bf16*)take((size_t)BB * NN * DD * 2);  // 64 MB
  __bf16* vT = (__bf16*)take((size_t)BB * NN * DD * 2);   // 64 MB

  prep_kernel<<<1, 256, 0, stream>>>(w0, b0, w1, b1, w2, b2, w3, b3, w4, b4,
                                     w5, b5, w7, b7, wk, wv, Whf, bh, Whp,
                                     wkp, wvp);
  initslots_kernel<<<1, 128, 0, stream>>>(mu, sg, slots);
  hkv_kernel<<<BB * NN / 128, 256, 0, stream>>>(inp, emb, Whp, bh, wkp, bk,
                                                wvp, bv, ling, linb, kbf, vT);
  for (int it = 0; it < 3; ++it) {
    qproj_kernel<<<8, 128, 0, stream>>>(slots, lsg, lsb, wq, bq, qTp);
    zero_kernel<<<64, 256, 0, stream>>>(U, R);
    attn_kernel<<<64, 128, 0, stream>>>(kbf, vT, qTp, U, R);
    finish_kernel<<<64, 128, 0, stream>>>(slots, U, R, wih, whh, bih, bhh,
                                          lfg, lfb, m1w, m1b, m2w, m2b, outp,
                                          (it == 2) ? 1 : 0);
  }
  (void)in_sizes; (void)n_in; (void)out_size; (void)ws_size;
}
